// SentenceReadout_10428180595138
// MI455X (gfx1250) — matrix operand order repaired and checked by reading, not compiled, not run on hardware
//
#include <hip/hip_runtime.h>
#include <hip/hip_bf16.h>

#define N_ROWS 8192
#define HDIM   256
#define NGRAPH 64
#define NSPLIT 2          // key-dimension split for flash attention
#define KBLK   (N_ROWS / 32 / NSPLIT)   // 32-key blocks per split = 128
#define XSTR   280        // LDS row stride in halves (bank-friendly, 16B aligned)
#define PSTR   40         // LDS row stride in halves for per-wave P scratch

typedef __attribute__((ext_vector_type(16))) _Float16 v16h;
typedef __attribute__((ext_vector_type(8)))  _Float16 v8h;
typedef __attribute__((ext_vector_type(8)))  float    v8f;

__device__ __forceinline__ v8f wmma_f16(const v16h a, const v16h b, const v8f c) {
  // D = A(16x32 f16) * B(32x16 f16) + C(16x16 f32)
  return __builtin_amdgcn_wmma_f32_16x16x32_f16(false, a, false, b, (short)0, c, false, false);
}

__device__ __forceinline__ v16h cat8(const v8h lo, const v8h hi) {
  return __builtin_shufflevector(lo, hi, 0, 1, 2, 3, 4, 5, 6, 7, 8, 9, 10, 11, 12, 13, 14, 15);
}

// ---------------------------------------------------------------------------
// Kernel 0: weights -> f16, transposed to [n][k] so B-fragments load contiguously
// ---------------------------------------------------------------------------
__global__ void prep_w_kernel(const float* __restrict__ We, const float* __restrict__ Wq,
                              const float* __restrict__ Wk, const float* __restrict__ Wv,
                              _Float16* __restrict__ wt) {
  const int gid = blockIdx.x * blockDim.x + threadIdx.x;   // 0 .. 4*H*H-1
  const int w = gid >> 16;
  const int r = gid & 65535;
  const int k = r >> 8;
  const int n = r & 255;
  const float* src = (w == 0) ? We : (w == 1) ? Wq : (w == 2) ? Wk : Wv;
  wt[(size_t)w * HDIM * HDIM + (size_t)n * HDIM + k] = (_Float16)src[(size_t)k * HDIM + n];
}

// ---------------------------------------------------------------------------
// 16x256 GEMM tile: A resident (8 frags from LDS), B streamed from global
// through a depth-3 ring so WMMA overlaps the global loads.
// ---------------------------------------------------------------------------
__device__ __forceinline__ void gemm_16x256(const _Float16* __restrict__ aLds,
                                            const _Float16* __restrict__ Bt,
                                            const int ln, const int hb, v8f acc[16]) {
  v16h af[8];
#pragma unroll
  for (int kc = 0; kc < 8; ++kc) {
    const v8h a0 = *(const v8h*)(aLds + ln * XSTR + kc * 32 + hb * 16);
    const v8h a1 = *(const v8h*)(aLds + ln * XSTR + kc * 32 + hb * 16 + 8);
    af[kc] = cat8(a0, a1);
  }
  const _Float16* bbase = Bt + (size_t)ln * HDIM + hb * 16;
#pragma unroll
  for (int nt = 0; nt < 16; ++nt) {
    const _Float16* bp = bbase + (size_t)nt * 16 * HDIM;   // row nt*16+ln, contiguous k
    v16h bf[4];
    bf[0] = *(const v16h*)(bp);
    bf[1] = *(const v16h*)(bp + 32);
    bf[2] = *(const v16h*)(bp + 64);
    v8f c;
#pragma unroll
    for (int v = 0; v < 8; ++v) c[v] = 0.0f;
#pragma unroll
    for (int kc = 0; kc < 8; ++kc) {
      if (kc + 3 < 8) bf[(kc + 3) & 3] = *(const v16h*)(bp + (kc + 3) * 32);
      c = wmma_f16(af[kc], bf[kc & 3], c);
    }
    acc[nt] = c;
  }
}

// ---------------------------------------------------------------------------
// Kernel 1: fused projections.  Block = 128 threads (4 waves), 64 rows.
//   h = relu(x@W_emb + b); Q = (h@Wq+bq)/16 (f16); K = h@Wk+bk (f16);
//   V stored transposed Vt[H][N] (f16) so attn's P@V B-fragment is contiguous.
// ---------------------------------------------------------------------------
__global__ __launch_bounds__(128) void proj_kernel(
    const float* __restrict__ x, const _Float16* __restrict__ wt,
    const float* __restrict__ b_emb, const float* __restrict__ bq,
    const float* __restrict__ bk, const float* __restrict__ bv,
    _Float16* __restrict__ Qs, _Float16* __restrict__ Kf, _Float16* __restrict__ Vt) {
  __shared__ _Float16 tile[64 * XSTR];   // reused: x tile, then h tile
  const int tid = threadIdx.x;
  const int w = tid >> 5;
  const int lane = tid & 31;
  const int ln = lane & 15;
  const int hb = lane >> 4;
  const int rowBlk = blockIdx.x * 64;

  // stage x tile (f32 -> f16) into LDS
#pragma unroll
  for (int it = 0; it < 32; ++it) {
    const int idx = it * 128 + tid;        // 64 rows * 64 float4 granules
    const int r = idx >> 6;
    const int c = (idx & 63) * 4;
    const float4 vx = *(const float4*)(x + (size_t)(rowBlk + r) * HDIM + c);
    _Float16* d = tile + r * XSTR + c;
    d[0] = (_Float16)vx.x; d[1] = (_Float16)vx.y;
    d[2] = (_Float16)vx.z; d[3] = (_Float16)vx.w;
  }
  __syncthreads();

  v8f acc[16];
  const _Float16* aBase = tile + w * 16 * XSTR;

  // h = relu(x @ W_emb + b_emb)
  gemm_16x256(aBase, wt, ln, hb, acc);
  __syncthreads();                         // all waves done reading x tile
#pragma unroll
  for (int nt = 0; nt < 16; ++nt) {
    const float b = b_emb[nt * 16 + ln];
#pragma unroll
    for (int v = 0; v < 8; ++v) {
      float h = acc[nt][v] + b;
      h = h > 0.0f ? h : 0.0f;
      // C/D layout: row = hb*8+v (within wave tile), col = nt*16+ln
      tile[(w * 16 + hb * 8 + v) * XSTR + nt * 16 + ln] = (_Float16)h;
    }
  }
  __syncthreads();

  // Q (pre-scaled by 1/sqrt(H) = 1/16, exact in f16)
  gemm_16x256(aBase, wt + 1 * HDIM * HDIM, ln, hb, acc);
#pragma unroll
  for (int nt = 0; nt < 16; ++nt) {
    const float b = bq[nt * 16 + ln];
    const int col = nt * 16 + ln;
#pragma unroll
    for (int v = 0; v < 8; ++v) {
      const int r = rowBlk + w * 16 + hb * 8 + v;
      Qs[(size_t)r * HDIM + col] = (_Float16)((acc[nt][v] + b) * 0.0625f);
    }
  }
  // K
  gemm_16x256(aBase, wt + 2 * HDIM * HDIM, ln, hb, acc);
#pragma unroll
  for (int nt = 0; nt < 16; ++nt) {
    const float b = bk[nt * 16 + ln];
    const int col = nt * 16 + ln;
#pragma unroll
    for (int v = 0; v < 8; ++v) {
      const int r = rowBlk + w * 16 + hb * 8 + v;
      Kf[(size_t)r * HDIM + col] = (_Float16)(acc[nt][v] + b);
    }
  }
  // V, stored transposed Vt[col][row]
  gemm_16x256(aBase, wt + 3 * HDIM * HDIM, ln, hb, acc);
#pragma unroll
  for (int nt = 0; nt < 16; ++nt) {
    const float b = bv[nt * 16 + ln];
    const int col = nt * 16 + ln;
#pragma unroll
    for (int v = 0; v < 8; ++v) {
      const int r = rowBlk + w * 16 + hb * 8 + v;
      Vt[(size_t)col * N_ROWS + r] = (_Float16)(acc[nt][v] + b);
    }
  }
}

// ---------------------------------------------------------------------------
// Kernel 2: flash attention, 2-way split over keys.  Each wave owns 16 query
// rows and one key half; online softmax; unnormalized O + (m, l) stored.
// ---------------------------------------------------------------------------
__global__ __launch_bounds__(128) void attn_kernel(
    const _Float16* __restrict__ Qs, const _Float16* __restrict__ Kf,
    const _Float16* __restrict__ Vt, float* __restrict__ Opart,
    float* __restrict__ ml) {
  __shared__ _Float16 ps[4][16 * PSTR];    // per-wave P repack scratch
  const int tid = threadIdx.x;
  const int w = tid >> 5;
  const int lane = tid & 31;
  const int ln = lane & 15;
  const int hb = lane >> 4;
  const int r0 = blockIdx.x * 64 + w * 16;
  const int split = blockIdx.y;
  float* O = Opart + (size_t)split * N_ROWS * HDIM;
  _Float16* myps = ps[w];

  // resident Q fragments (16 rows x 256 k)
  v16h qf[8];
#pragma unroll
  for (int kc = 0; kc < 8; ++kc) {
    const _Float16* qp = Qs + (size_t)(r0 + ln) * HDIM + kc * 32 + hb * 16;
    qf[kc] = cat8(*(const v8h*)qp, *(const v8h*)(qp + 8));
  }

  v8f o[16];
#pragma unroll
  for (int nt = 0; nt < 16; ++nt)
#pragma unroll
    for (int v = 0; v < 8; ++v) o[nt][v] = 0.0f;

  float mrun[8], lsum[8];
#pragma unroll
  for (int v = 0; v < 8; ++v) { mrun[v] = -3.0e38f; lsum[v] = 0.0f; }

  const int jend = (split + 1) * KBLK;
  for (int j = split * KBLK; j < jend; ++j) {
    const int kb = j * 32;
    if (j + 1 < jend)
      __builtin_prefetch(Kf + (size_t)(kb + 32 + lane) * HDIM, 0, 1);

    // ---- S = Qs @ K^T : 16-load sequence through a 4-slot ring -------------
    // seq s: kc = s>>1, half = s&1 (row kb+ln or kb+16+ln)
    const _Float16* kbase = Kf + (size_t)(kb + ln) * HDIM + hb * 16;
    v8f s0, s1;
#pragma unroll
    for (int v = 0; v < 8; ++v) { s0[v] = 0.0f; s1[v] = 0.0f; }
    v16h kf[4];
    kf[0] = *(const v16h*)(kbase);                       // s=0: kc0, half0
    kf[1] = *(const v16h*)(kbase + 16 * HDIM);           // s=1: kc0, half1
    kf[2] = *(const v16h*)(kbase + 32);                  // s=2: kc1, half0
#pragma unroll
    for (int s = 0; s < 16; ++s) {
      if (s + 3 < 16) {
        const int sn = s + 3;
        kf[sn & 3] = *(const v16h*)(kbase + (sn & 1) * 16 * HDIM + (sn >> 1) * 32);
      }
      if (s & 1) s1 = wmma_f16(qf[s >> 1], kf[s & 3], s1);
      else       s0 = wmma_f16(qf[s >> 1], kf[s & 3], s0);
    }

    // ---- start V loads early (independent of softmax) ----------------------
    const _Float16* vbase = Vt + (size_t)ln * N_ROWS + kb + hb * 16;
    v16h vb[4];
    vb[0] = *(const v16h*)(vbase);
    vb[1] = *(const v16h*)(vbase + (size_t)16 * N_ROWS);
    vb[2] = *(const v16h*)(vbase + (size_t)32 * N_ROWS);

    // ---- online softmax (row = hb*8+v striped across 16 lanes) -------------
    float t[8];
#pragma unroll
    for (int v = 0; v < 8; ++v) t[v] = fmaxf(s0[v], s1[v]);
#pragma unroll
    for (int d = 1; d < 16; d <<= 1)
#pragma unroll
      for (int v = 0; v < 8; ++v) t[v] = fmaxf(t[v], __shfl_xor(t[v], d, 32));

    float corr[8];
#pragma unroll
    for (int v = 0; v < 8; ++v) {
      const float mn = fmaxf(mrun[v], t[v]);
      corr[v] = __expf(mrun[v] - mn);
      mrun[v] = mn;
      lsum[v] *= corr[v];
    }
#pragma unroll
    for (int nt = 0; nt < 16; ++nt)
#pragma unroll
      for (int v = 0; v < 8; ++v) o[nt][v] *= corr[v];

    // P = exp(S - m); stash in LDS (C-layout write), reload in A-layout
#pragma unroll
    for (int v = 0; v < 8; ++v) {
      const float p0 = __expf(s0[v] - mrun[v]);
      const float p1 = __expf(s1[v] - mrun[v]);
      lsum[v] += p0 + p1;
      myps[(hb * 8 + v) * PSTR + ln]      = (_Float16)p0;
      myps[(hb * 8 + v) * PSTR + 16 + ln] = (_Float16)p1;
    }
    asm volatile("s_wait_dscnt 0" ::: "memory");   // in-wave LDS store->load ordering
    const v16h pa = cat8(*(const v8h*)(myps + ln * PSTR + hb * 16),
                         *(const v8h*)(myps + ln * PSTR + hb * 16 + 8));

    // ---- O += P @ V : 16 fragments through a 4-slot ring --------------------
#pragma unroll
    for (int nt = 0; nt < 16; ++nt) {
      if (nt + 3 < 16)
        vb[(nt + 3) & 3] = *(const v16h*)(vbase + (size_t)(nt + 3) * 16 * N_ROWS);
      o[nt] = wmma_f16(pa, vb[nt & 3], o[nt]);
    }
  }

  // finalize this split: reduce lsum across 16 lanes; store unnormalized O, m, l
#pragma unroll
  for (int d = 1; d < 16; d <<= 1)
#pragma unroll
    for (int v = 0; v < 8; ++v) lsum[v] += __shfl_xor(lsum[v], d, 32);
#pragma unroll
  for (int nt = 0; nt < 16; ++nt)
#pragma unroll
    for (int v = 0; v < 8; ++v)
      O[(size_t)(r0 + hb * 8 + v) * HDIM + nt * 16 + ln] = o[nt][v];
  if (ln == 0) {
#pragma unroll
    for (int v = 0; v < 8; ++v) {
      const int r = r0 + hb * 8 + v;
      ml[(size_t)split * N_ROWS + r]                    = mrun[v];  // m
      ml[(size_t)(NSPLIT + split) * N_ROWS + r]         = lsum[v];  // l
    }
  }
}

// ---------------------------------------------------------------------------
// Kernel 2b: merge the two key-splits; writes normalized result into O0.
// ---------------------------------------------------------------------------
__global__ __launch_bounds__(HDIM) void merge_kernel(float* __restrict__ O0,
                                                     const float* __restrict__ O1,
                                                     const float* __restrict__ ml) {
  const int row = blockIdx.x;
  const int h = threadIdx.x;
  const float m0 = ml[row];
  const float m1 = ml[N_ROWS + row];
  const float l0 = ml[2 * N_ROWS + row];
  const float l1 = ml[3 * N_ROWS + row];
  const float ms = fmaxf(m0, m1);
  const float f0 = __expf(m0 - ms);
  const float f1 = __expf(m1 - ms);
  const float inv = 1.0f / (l0 * f0 + l1 * f1);
  const size_t idx = (size_t)row * HDIM + h;
  O0[idx] = (O0[idx] * f0 + O1[idx] * f1) * inv;
}

// ---------------------------------------------------------------------------
// Kernel 3: segment starts via binary search (batch ids are sorted)
// ---------------------------------------------------------------------------
__global__ void seg_starts_kernel(const int* __restrict__ batch, int* __restrict__ starts) {
  const int b = threadIdx.x;
  if (b > NGRAPH) return;
  int lo = 0, hi = N_ROWS;
  while (lo < hi) {
    const int mid = (lo + hi) >> 1;
    if (batch[mid] < b) lo = mid + 1; else hi = mid;
  }
  starts[b] = lo;
}

// ---------------------------------------------------------------------------
// Kernel 4: per-graph max/mean pooling -> out[B, 2H] = [max | mean]
// ---------------------------------------------------------------------------
__global__ __launch_bounds__(HDIM) void pool_kernel(const float* __restrict__ att,
                                                    const int* __restrict__ starts,
                                                    float* __restrict__ out) {
  const int b = blockIdx.x;
  const int h = threadIdx.x;
  const int s = starts[b], e = starts[b + 1];
  float mx = 0.0f, sm = 0.0f;
  if (s < e) {
    mx = -3.4e38f;
    for (int i = s; i < e; ++i) {
      const float v = att[(size_t)i * HDIM + h];
      mx = fmaxf(mx, v);
      sm += v;
    }
    sm /= (float)(e - s);
  }
  out[(size_t)b * 2 * HDIM + h] = mx;
  out[(size_t)b * 2 * HDIM + HDIM + h] = sm;
}

// ---------------------------------------------------------------------------
extern "C" void kernel_launch(void* const* d_in, const int* in_sizes, int n_in,
                              void* d_out, int out_size, void* d_ws, size_t ws_size,
                              hipStream_t stream) {
  (void)in_sizes; (void)n_in; (void)out_size; (void)ws_size;
  const float* x     = (const float*)d_in[0];
  const float* W_emb = (const float*)d_in[1];
  const float* b_emb = (const float*)d_in[2];
  const float* Wq    = (const float*)d_in[3];
  const float* bq    = (const float*)d_in[4];
  const float* Wk    = (const float*)d_in[5];
  const float* bk    = (const float*)d_in[6];
  const float* Wv    = (const float*)d_in[7];
  const float* bv    = (const float*)d_in[8];
  const int*   batch = (const int*)d_in[9];
  float* out = (float*)d_out;

  char* ws = (char*)d_ws;
  _Float16* wt   = (_Float16*)(ws);                          // 4*H*H f16 = 512 KiB
  _Float16* Qs   = (_Float16*)(ws + 524288);                 // N*H f16   = 4 MiB
  _Float16* Kf   = (_Float16*)(ws + 4718592);                // N*H f16
  _Float16* Vt   = (_Float16*)(ws + 8912896);                // H*N f16 (transposed)
  float*    O01  = (float*)   (ws + 13107200);               // 2 * N*H f32 = 16 MiB
  float*    ml   = (float*)   (ws + 29884416);               // 4*N f32 (m0,m1,l0,l1)
  int*   starts  = (int*)     (ws + 30015488);               // B+1 ints

  float* O0 = O01;
  float* O1 = O01 + (size_t)N_ROWS * HDIM;

  prep_w_kernel<<<(4 * HDIM * HDIM) / 256, 256, 0, stream>>>(W_emb, Wq, Wk, Wv, wt);
  proj_kernel<<<N_ROWS / 64, 128, 0, stream>>>(x, wt, b_emb, bq, bk, bv, Qs, Kf, Vt);
  attn_kernel<<<dim3(N_ROWS / 64, NSPLIT), 128, 0, stream>>>(Qs, Kf, Vt, O01, ml);
  merge_kernel<<<N_ROWS, HDIM, 0, stream>>>(O0, O1, ml);
  seg_starts_kernel<<<1, 96, 0, stream>>>(batch, starts);
  pool_kernel<<<NGRAPH, HDIM, 0, stream>>>(O0, starts, out);
}
